// Generator_LSTM_17317308137811
// MI455X (gfx1250) — matrix operand order repaired and checked by reading, not compiled, not run on hardware
//
#include <hip/hip_runtime.h>
#include <hip/hip_bf16.h>
#include <math.h>

#define B_  1024
#define V_  256
#define H_  1536
#define T_  256
#define G4_ (4 * H_)   // 6144 gates

#define BLK_M 128
#define BLK_N 128
#define LDA_S 40       // LDS row stride in bf16 (80B: 16B-aligned, bank-conflict-free)

typedef __attribute__((ext_vector_type(16))) __bf16 v16bf;
typedef __attribute__((ext_vector_type(8)))  __bf16 v8bf;
typedef __attribute__((ext_vector_type(8)))  float  v8f;

static __device__ __forceinline__ __bf16 f2bf(float f) {
    union { unsigned short s; __bf16 b; } u;
    unsigned x = __float_as_uint(f);
    x += 0x7fffu + ((x >> 16) & 1u);           // round-to-nearest-even
    u.s = (unsigned short)(x >> 16);
    return u.b;
}

// ---------------------------------------------------------------------------
// bf16 WMMA GEMM:  out[M,N] = A[M,K](bf16,lda) * W[N,K]^T(bf16)
//                             + bias[N] + addf[M,N] + gatherW[n*V_ + gidx[m]]
// Block tile 128x128 (8 waves = 4m x 2n), wave tile 32x64 (2x4 WMMA 16x16x32).
// A tile staged through double-buffered LDS (coalesced b128 global loads,
// broadcast to all n-waves); W fragments load contiguously from global/L2.
// M % 128 == 0, N % 128 == 0, K % 32 == 0.
// ---------------------------------------------------------------------------
__global__ __launch_bounds__(256)
void gemm_bf16_wmma(const __bf16* __restrict__ A, int lda,
                    const __bf16* __restrict__ W, int K,
                    float* __restrict__ out, int N,
                    const float* __restrict__ bias,
                    const float* __restrict__ addf,
                    const float* __restrict__ gatherW,
                    const int*   __restrict__ gidx)
{
    __shared__ __bf16 smA[2][BLK_M * LDA_S];   // 2 x 10 KB

    const int tid  = threadIdx.x;
    const int lane = tid & 31;
    const int wave = tid >> 5;
    const int l15  = lane & 15;
    const int lh   = lane >> 4;
    const int wm   = wave >> 1;                 // 0..3
    const int wn   = wave & 1;                  // 0..1
    const int mBlock = blockIdx.x * BLK_M;
    const int mBase  = wm * 32;                 // within block
    const int nBase  = blockIdx.y * BLK_N + wn * 64;

    v8f acc[2][4];
    const v8f vzero = {0.f, 0.f, 0.f, 0.f, 0.f, 0.f, 0.f, 0.f};
#pragma unroll
    for (int i = 0; i < 2; ++i)
#pragma unroll
        for (int j = 0; j < 4; ++j) acc[i][j] = vzero;

    // A staging: 256 threads cover 128 rows x 32 k (2 chunks of 16B each)
    const int arow = tid >> 2;                  // 0..63
    const int acol = (tid & 3) * 8;             // 0,8,16,24 (bf16 elems)
    const __bf16* Ag0 = A + (size_t)(mBlock + arow)      * lda + acol;
    const __bf16* Ag1 = A + (size_t)(mBlock + 64 + arow) * lda + acol;

    const __bf16* Wptr[4];
#pragma unroll
    for (int t = 0; t < 4; ++t)
        Wptr[t] = W + (size_t)(nBase + t * 16 + l15) * K;

    v8bf areg0 = *reinterpret_cast<const v8bf*>(Ag0);
    v8bf areg1 = *reinterpret_cast<const v8bf*>(Ag1);

    for (int k = 0; k < K; k += 32) {
        const int stage = (k >> 5) & 1;
        // stage A tile into LDS
        *reinterpret_cast<v8bf*>(&smA[stage][arow * LDA_S + acol])        = areg0;
        *reinterpret_cast<v8bf*>(&smA[stage][(64 + arow) * LDA_S + acol]) = areg1;
        if (k + 32 < K) {                       // prefetch next k-slab to regs
            areg0 = *reinterpret_cast<const v8bf*>(Ag0 + k + 32);
            areg1 = *reinterpret_cast<const v8bf*>(Ag1 + k + 32);
        }
        __syncthreads();

        // B fragments: one contiguous 32B chunk per lane from W row n
        const int kb = k + lh * 16;
        v16bf b[4];
#pragma unroll
        for (int nt = 0; nt < 4; ++nt) {
            b[nt] = *reinterpret_cast<const v16bf*>(Wptr[nt] + kb);
            __builtin_prefetch(Wptr[nt] + kb + 64, 0, 1);   // global_prefetch_b8
        }
        // A fragments from LDS: K groups [lh*16 .. +7] and [lh*16+8 .. +15]
        union { v16bf v; v8bf h[2]; } a[2];
#pragma unroll
        for (int mt = 0; mt < 2; ++mt) {
            const int row = mBase + mt * 16 + l15;
            a[mt].h[0] = *reinterpret_cast<const v8bf*>(&smA[stage][row * LDA_S + lh * 16]);
            a[mt].h[1] = *reinterpret_cast<const v8bf*>(&smA[stage][row * LDA_S + lh * 16 + 8]);
        }
#pragma unroll
        for (int mt = 0; mt < 2; ++mt)
#pragma unroll
            for (int nt = 0; nt < 4; ++nt)
                acc[mt][nt] = __builtin_amdgcn_wmma_f32_16x16x32_bf16(
                    false, a[mt].v, false, b[nt], (short)0, acc[mt][nt],
                    false, false);
    }

    // Epilogue. C layout: VGPR v -> m = v + 8*(lane>>4); n = lane&15.
#pragma unroll
    for (int mt = 0; mt < 2; ++mt) {
#pragma unroll
        for (int nt = 0; nt < 4; ++nt) {
            const int n = nBase + nt * 16 + l15;
            float bn = bias ? bias[n] : 0.0f;
#pragma unroll
            for (int v = 0; v < 8; ++v) {
                const int m = mBlock + mBase + mt * 16 + lh * 8 + v;
                float r = acc[mt][nt][v] + bn;
                if (addf)    r += addf[(size_t)m * N + n];
                if (gatherW) r += gatherW[(size_t)n * V_ + gidx[m]];
                out[(size_t)m * N + n] = r;
            }
        }
    }
}

// ---------------------------------------------------------------------------
// Elementwise LSTM gate activation: gates[B,4H] (i,f,g,o) + c[B,H] ->
// c_new (fp32, in place) and h_new written as bf16 into hdst[b*ldh + hoff + j]
// ---------------------------------------------------------------------------
__global__ __launch_bounds__(256)
void lstm_act_kernel(const float* __restrict__ gates,
                     float* __restrict__ c,
                     __bf16* __restrict__ hdst, int ldh, int hoff)
{
    const int i = blockIdx.x * 256 + threadIdx.x;   // 0 .. B*H-1
    const int b = i / H_;
    const int j = i - b * H_;
    const float* g = gates + (size_t)b * G4_;
    const float gi = g[j];
    const float gf = g[H_ + j];
    const float gg = g[2 * H_ + j];
    const float go = g[3 * H_ + j];
    const float si = 1.0f / (1.0f + __expf(-gi));
    const float sf = 1.0f / (1.0f + __expf(-gf));
    const float so = 1.0f / (1.0f + __expf(-go));
    const float cn = sf * c[i] + si * tanhf(gg);
    c[i] = cn;
    const float hn = so * tanhf(cn);
    hdst[(size_t)b * ldh + hoff + j] = f2bf(hn);
}

// ---------------------------------------------------------------------------
// Gumbel-max categorical sampling, one wave32 per batch row (V=256 logits).
// ---------------------------------------------------------------------------
static __device__ __forceinline__ unsigned pcg(unsigned x) {
    x = x * 747796405u + 2891336453u;
    x = ((x >> ((x >> 28) + 4u)) ^ x) * 277803737u;
    return (x >> 22) ^ x;
}

__global__ __launch_bounds__(256)
void sample_kernel(const float* __restrict__ logits,
                   const int* __restrict__ seqlen,
                   int t, int* __restrict__ outT, int* __restrict__ idxbuf)
{
    const int lane = threadIdx.x & 31;
    const int row  = blockIdx.x * 8 + (threadIdx.x >> 5);
    const float* lr = logits + (size_t)row * V_;
    float best = -1e30f;
    int   bi = 0;
#pragma unroll
    for (int i = 0; i < 8; ++i) {
        const int n = i * 32 + lane;
        const unsigned h = pcg((unsigned)(t * 1048573 + row * 257 + n) ^ 0x9e3779b9u);
        const float u = (float)(h >> 8) * (1.0f / 16777216.0f) + 1e-10f;
        const float gmb = -__logf(-__logf(u));
        const float v = lr[n] + gmb;
        if (v > best) { best = v; bi = n; }
    }
#pragma unroll
    for (int off = 16; off > 0; off >>= 1) {
        const float ov = __shfl_xor(best, off, 32);
        const int   oi = __shfl_xor(bi,   off, 32);
        if (ov > best || (ov == best && oi < bi)) { best = ov; bi = oi; }
    }
    if (lane == 0) {
        idxbuf[row] = bi;
        outT[(size_t)t * B_ + row] = (t < seqlen[row]) ? bi : 0;
    }
}

// ---------------------------------------------------------------------------
// Utility kernels
// ---------------------------------------------------------------------------
__global__ __launch_bounds__(256)
void f32_to_bf16_kernel(const float* __restrict__ src, __bf16* __restrict__ dst, unsigned n)
{
    const unsigned i = blockIdx.x * 256u + threadIdx.x;
    if (i < n) dst[i] = f2bf(src[i]);
}

// interleave into [N, 2H] : dst[r*ldd + off + c] = bf16(src[r*rowlen + c])
__global__ __launch_bounds__(256)
void f32_to_bf16_cat_kernel(const float* __restrict__ src, __bf16* __restrict__ dst,
                            int rowlen, int ldd, int off, unsigned n)
{
    const unsigned i = blockIdx.x * 256u + threadIdx.x;
    if (i < n) {
        const unsigned r = i / (unsigned)rowlen;
        const unsigned c = i - r * (unsigned)rowlen;
        dst[(size_t)r * ldd + off + c] = f2bf(src[i]);
    }
}

__global__ __launch_bounds__(256)
void bias_sum_kernel(const float* __restrict__ a, const float* __restrict__ b,
                     float* __restrict__ o, unsigned n)
{
    const unsigned i = blockIdx.x * 256u + threadIdx.x;
    if (i < n) o[i] = a[i] + b[i];
}

__global__ __launch_bounds__(256)
void zero_u32_kernel(unsigned* __restrict__ p, unsigned n)
{
    const unsigned i = blockIdx.x * 256u + threadIdx.x;
    if (i < n) p[i] = 0u;
}

// ---------------------------------------------------------------------------
extern "C" void kernel_launch(void* const* d_in, const int* in_sizes, int n_in,
                              void* d_out, int out_size, void* d_ws, size_t ws_size,
                              hipStream_t stream)
{
    (void)in_sizes; (void)n_in; (void)out_size; (void)ws_size;
    const float* noise  = (const float*)d_in[0];
    const int*   seqlen = (const int*)  d_in[1];
    const float* W_ih0  = (const float*)d_in[2];
    const float* W_hh0  = (const float*)d_in[3];
    const float* b_ih0  = (const float*)d_in[4];
    const float* b_hh0  = (const float*)d_in[5];
    const float* W_ih1  = (const float*)d_in[6];
    const float* W_hh1  = (const float*)d_in[7];
    const float* b_ih1  = (const float*)d_in[8];
    const float* b_hh1  = (const float*)d_in[9];
    const float* W_out  = (const float*)d_in[10];
    const float* b_out  = (const float*)d_in[11];
    int* out = (int*)d_out;

    // Workspace layout (256-byte aligned chunks)
    char* ws = (char*)d_ws;
    size_t off = 0;
    auto alloc = [&](size_t bytes) -> char* {
        char* p = ws + off;
        off += (bytes + 255) & ~(size_t)255;
        return p;
    };
    __bf16* Whh0_bf  = (__bf16*)alloc((size_t)G4_ * H_ * 2);      // [6144,1536]
    __bf16* Wcat1_bf = (__bf16*)alloc((size_t)G4_ * 2 * H_ * 2);  // [6144,3072] = [Wih1|Whh1]
    __bf16* Wout_bf  = (__bf16*)alloc((size_t)V_ * H_ * 2);       // [256,1536]
    __bf16* Wih0_bf  = (__bf16*)alloc((size_t)G4_ * V_ * 2);      // [6144,256]
    __bf16* noise_bf = (__bf16*)alloc((size_t)B_ * V_ * 2);       // [1024,256]
    __bf16* hcat     = (__bf16*)alloc((size_t)B_ * 2 * H_ * 2);   // [1024,3072] = [h0|h1]
    float*  xw0      = (float*) alloc((size_t)B_ * G4_ * 4);      // t=0 input term
    float*  gates    = (float*) alloc((size_t)B_ * G4_ * 4);
    float*  c0       = (float*) alloc((size_t)B_ * H_ * 4);
    float*  c1       = (float*) alloc((size_t)B_ * H_ * 4);
    float*  logits   = (float*) alloc((size_t)B_ * V_ * 4);
    float*  bias0    = (float*) alloc(G4_ * 4);
    float*  bias1    = (float*) alloc(G4_ * 4);
    int*    idxbuf   = (int*)   alloc(B_ * 4);

    const dim3 blk(256);
    auto gr = [](unsigned n) { return dim3((n + 255u) / 256u); };

    // ---- init state (hcat is B*2H bf16 = B*H dwords) ----
    zero_u32_kernel<<<gr(B_ * H_), blk, 0, stream>>>((unsigned*)c0, B_ * H_);
    zero_u32_kernel<<<gr(B_ * H_), blk, 0, stream>>>((unsigned*)c1, B_ * H_);
    zero_u32_kernel<<<gr(B_ * H_), blk, 0, stream>>>((unsigned*)hcat, B_ * H_);

    // ---- weight conversion (every call: deterministic, no caching) ----
    f32_to_bf16_kernel<<<gr(G4_ * H_), blk, 0, stream>>>(W_hh0, Whh0_bf, G4_ * H_);
    f32_to_bf16_cat_kernel<<<gr(G4_ * H_), blk, 0, stream>>>(W_ih1, Wcat1_bf, H_, 2 * H_, 0,  G4_ * H_);
    f32_to_bf16_cat_kernel<<<gr(G4_ * H_), blk, 0, stream>>>(W_hh1, Wcat1_bf, H_, 2 * H_, H_, G4_ * H_);
    f32_to_bf16_kernel<<<gr(V_ * H_), blk, 0, stream>>>(W_out, Wout_bf, V_ * H_);
    f32_to_bf16_kernel<<<gr(G4_ * V_), blk, 0, stream>>>(W_ih0, Wih0_bf, G4_ * V_);
    f32_to_bf16_kernel<<<gr(B_ * V_), blk, 0, stream>>>(noise, noise_bf, B_ * V_);
    bias_sum_kernel<<<gr(G4_), blk, 0, stream>>>(b_ih0, b_hh0, bias0, G4_);
    bias_sum_kernel<<<gr(G4_), blk, 0, stream>>>(b_ih1, b_hh1, bias1, G4_);

    // ---- t=0 dense input term: xw0 = noise @ W_ih0^T  [1024 x 6144, K=256] ----
    gemm_bf16_wmma<<<dim3(B_ / BLK_M, G4_ / BLK_N), blk, 0, stream>>>(
        noise_bf, V_, Wih0_bf, V_, xw0, G4_, nullptr, nullptr, nullptr, nullptr);

    const dim3 gBig(B_ / BLK_M, G4_ / BLK_N);  // (8, 48)
    const dim3 gOut(B_ / BLK_M, V_ / BLK_N);   // (8, 2)
    const dim3 gAct(B_ * H_ / 256);

    for (int t = 0; t < T_; ++t) {
        // Layer 0: gates = h0 @ W_hh0^T + bias + (t==0 ? noise@Wih0^T : Wih0[:,idx])
        gemm_bf16_wmma<<<gBig, blk, 0, stream>>>(
            hcat, 2 * H_, Whh0_bf, H_, gates, G4_,
            bias0,
            (t == 0) ? xw0 : nullptr,
            (t > 0) ? W_ih0 : nullptr, idxbuf);
        lstm_act_kernel<<<gAct, blk, 0, stream>>>(gates, c0, hcat, 2 * H_, 0);

        // Layer 1: gates = [h0|h1] @ [Wih1|Whh1]^T + bias   (K = 3072)
        gemm_bf16_wmma<<<gBig, blk, 0, stream>>>(
            hcat, 2 * H_, Wcat1_bf, 2 * H_, gates, G4_,
            bias1, nullptr, nullptr, nullptr);
        lstm_act_kernel<<<gAct, blk, 0, stream>>>(gates, c1, hcat, 2 * H_, H_);

        // Output projection: logits = h1 @ W_out^T + b_out  [1024 x 256, K=1536]
        gemm_bf16_wmma<<<gOut, blk, 0, stream>>>(
            hcat + H_, 2 * H_, Wout_bf, H_, logits, V_,
            b_out, nullptr, nullptr, nullptr);

        // Gumbel-max categorical sample -> idxbuf, out[t, :]
        sample_kernel<<<dim3(B_ / 8), blk, 0, stream>>>(logits, seqlen, t, out, idxbuf);
    }
}
